// RobustKMeansQuantizer_65884798320943
// MI455X (gfx1250) — matrix operand order repaired and checked by reading, not compiled, not run on hardware
//
#include <hip/hip_runtime.h>
#include <hip/hip_bf16.h>

// Problem constants (from reference)
#define BB 16
#define LL 8192
#define DCH 256
#define KC 320
#define NROWS (BB * LL)          // 131072
#define CODES_ELEMS ((long)NROWS * DCH)  // 33554432

// ws layout (float offsets)
#define SUM_OFF   0      // 256
#define SSQ_OFF   256    // 256
#define SCALE_OFF 512    // 256
#define SHIFT_OFF 768    // 256
#define CNORM_OFF 1024   // 320  ( -0.5*||c||^2 )
#define CBH_OFF   1536   // f16 codebook, 320*256 halves = 40960 float slots

#define PITCH_DW 132     // LDS dwords per codebook row: 16B-aligned, bank-benign for b128

typedef __attribute__((ext_vector_type(16))) _Float16 v16h;
typedef __attribute__((ext_vector_type(8)))  float    v8f;

__global__ void kq_zero(float* ws) {
    ws[blockIdx.x * 256 + threadIdx.x] = 0.0f;
}

// Per-channel sum / sumsq. 1024 blocks x 256 threads, 128 rows per block.
__global__ __launch_bounds__(256) void kq_stats(const float* __restrict__ x,
                                                float* __restrict__ ws) {
    const int ch = threadIdx.x;
    const long base = (long)blockIdx.x * 128 * DCH;
    float s = 0.0f, q = 0.0f;
    #pragma unroll 4
    for (int r = 0; r < 128; ++r) {
        float v = x[base + (long)r * DCH + ch];
        s += v;
        q += v * v;
    }
    atomicAdd(&ws[SUM_OFF + ch], s);
    atomicAdd(&ws[SSQ_OFF + ch], q);
}

// Fold stats into per-channel affine: xb = x*scale + shift. 1 block x 256.
__global__ void kq_prep_bn(const float* __restrict__ w, const float* __restrict__ b,
                           float* __restrict__ ws) {
    const int ch = threadIdx.x;
    const float invN = 1.0f / (float)NROWS;
    float mean = ws[SUM_OFF + ch] * invN;
    float var  = ws[SSQ_OFF + ch] * invN - mean * mean;
    float sc   = w[ch] * rsqrtf(var + 1e-5f);
    ws[SCALE_OFF + ch] = sc;
    ws[SHIFT_OFF + ch] = b[ch] - mean * sc;
}

// Convert codebook to f16 + compute -0.5*||c||^2. 320 blocks x 256 threads.
__global__ __launch_bounds__(256) void kq_prep_cb(const float* __restrict__ cb,
                                                  float* __restrict__ ws) {
    const int k = blockIdx.x, t = threadIdx.x;
    float v = cb[(long)k * DCH + t];
    _Float16* cbh = (_Float16*)(ws + CBH_OFF);
    cbh[(long)k * DCH + t] = (_Float16)v;
    float q = v * v;
    for (int m = 16; m; m >>= 1) q += __shfl_down(q, m, 32);
    __shared__ float red[8];
    if ((t & 31) == 0) red[t >> 5] = q;
    __syncthreads();
    if (t == 0) {
        float tot = 0.0f;
        #pragma unroll
        for (int i = 0; i < 8; ++i) tot += red[i];
        ws[CNORM_OFF + k] = -0.5f * tot;
    }
}

// Main fused kernel: normalize -> WMMA scores vs codebook -> argmax -> write idx + gather codes.
// Grid: 1024 blocks x 256 threads (8 waves, 16 rows per wave, 128 rows per block).
// Dynamic LDS: f16 codebook, 320 rows x PITCH_DW dwords = 168,960 bytes.
__global__ __launch_bounds__(256) void kq_main(const float* __restrict__ x,
                                               const float* __restrict__ cbf,
                                               const float* __restrict__ ws,
                                               float* __restrict__ out) {
    extern __shared__ unsigned int lds[];
    const unsigned int* cbh = (const unsigned int*)(ws + CBH_OFF); // 320*128 dwords

    // Stage f16 codebook into LDS with 16B-aligned padded pitch.
    for (int i = threadIdx.x; i < KC * 128; i += 256) {
        int row = i >> 7, col = i & 127;
        lds[row * PITCH_DW + col] = cbh[i];
    }
    __syncthreads();

    const int lane = threadIdx.x & 31;
    const int wave = threadIdx.x >> 5;
    const int h = lane >> 4;        // half-wave selector
    const int n = lane & 15;        // A: row-in-tile; B/C: column (code) in tile
    const long rowBase = (long)blockIdx.x * 128 + wave * 16;
    const long myRow = rowBase + n;

    const float* scale = ws + SCALE_OFF;
    const float* shift = ws + SHIFT_OFF;
    const float* cnorm = ws + CNORM_OFF;

    // Load + normalize one 16x256 row tile as 8 f16 A-fragments (16x32 each).
    // 16-bit A 16x32 layout: K(j,h) = j*2 + h*16 (contiguous, matches B).
    v16h a[8];
    const float* xr = x + myRow * DCH;
    #pragma unroll
    for (int dd = 0; dd < 8; ++dd) {
        #pragma unroll
        for (int j = 0; j < 8; ++j) {
            int k = dd * 32 + j * 2 + h * 16;
            float v0 = xr[k]     * scale[k]     + shift[k];
            float v1 = xr[k + 1] * scale[k + 1] + shift[k + 1];
            a[dd][2 * j]     = (_Float16)v0;
            a[dd][2 * j + 1] = (_Float16)v1;
        }
    }

    float bestv[8];
    int   besti[8];
    #pragma unroll
    for (int r = 0; r < 8; ++r) { bestv[r] = -3.0e38f; besti[r] = 0; }

    const uint4* lds4 = (const uint4*)lds;

    // 20 code tiles of 16; score s = xb.c - 0.5*||c||^2 (argmax == argmin of d2).
    // Fully unrolled so the scheduler can hoist B-fragment/cnorm loads across tiles.
    #pragma unroll
    for (int kk = 0; kk < 20; ++kk) {
        const int code = kk * 16 + n;
        const float cn = cnorm[code];   // off the WMMA chain; folded in at compare time
        v8f c = {};                     // C starts at inline 0

        #pragma unroll
        for (int dd = 0; dd < 8; ++dd) {
            // 16-bit B 32x16 layout: lane column n; dwords j=0..7 contiguous at
            // code*PITCH_DW + dd*16 + h*8  -> two ds_load_b128 per fragment.
            union { v16h v; uint4 q[2]; } bfrag;
            const int base4 = (code * PITCH_DW + dd * 16 + h * 8) >> 2;
            bfrag.q[0] = lds4[base4];
            bfrag.q[1] = lds4[base4 + 1];
            c = __builtin_amdgcn_wmma_f32_16x16x32_f16(
                    false, a[dd], false, bfrag.v, (short)0, c, false, false);
        }
        // C layout: VGPR r -> row (r + 8*h), column n -> candidate index = code.
        #pragma unroll
        for (int r = 0; r < 8; ++r) {
            float s = c[r] + cn;
            if (s > bestv[r]) { bestv[r] = s; besti[r] = code; }
        }
    }

    // Butterfly (val,idx)-max across the 16 lanes of each half-wave; ties -> lower index
    // (matches argmin-first semantics).
    #pragma unroll
    for (int m = 1; m < 16; m <<= 1) {
        #pragma unroll
        for (int r = 0; r < 8; ++r) {
            float ov = __shfl_xor(bestv[r], m, 32);
            int   oi = __shfl_xor(besti[r], m, 32);
            if (ov > bestv[r] || (ov == bestv[r] && oi < besti[r])) {
                bestv[r] = ov; besti[r] = oi;
            }
        }
    }

    // Write indices (as float values) into the tail of d_out.
    float* idxOut = out + CODES_ELEMS;
    if (n == 0) {
        #pragma unroll
        for (int r = 0; r < 8; ++r)
            idxOut[rowBase + h * 8 + r] = (float)besti[r];
    }

    // Gather winning codebook rows (f32, L2-resident) into the codes region.
    #pragma unroll
    for (int m = 0; m < 16; ++m) {
        int idx = __shfl(besti[m & 7], (m >> 3) << 4, 32);
        const float4* src = (const float4*)(cbf + (long)idx * DCH);
        float4* dst = (float4*)(out + (rowBase + m) * DCH);
        dst[lane]      = src[lane];        // 64 float4 per row, 2 per lane
        dst[lane + 32] = src[lane + 32];
    }
}

extern "C" void kernel_launch(void* const* d_in, const int* in_sizes, int n_in,
                              void* d_out, int out_size, void* d_ws, size_t ws_size,
                              hipStream_t stream) {
    const float* x   = (const float*)d_in[0];
    const float* bnw = (const float*)d_in[1];
    const float* bnb = (const float*)d_in[2];
    const float* cb  = (const float*)d_in[3];
    float* out = (float*)d_out;
    float* ws  = (float*)d_ws;

    kq_zero<<<2, 256, 0, stream>>>(ws);
    kq_stats<<<NROWS / 128, 256, 0, stream>>>(x, ws);
    kq_prep_bn<<<1, 256, 0, stream>>>(bnw, bnb, ws);
    kq_prep_cb<<<KC, 256, 0, stream>>>(cb, ws);

    const size_t ldsBytes = (size_t)KC * PITCH_DW * 4;  // 168,960 B < 320 KB/WGP
    kq_main<<<NROWS / 128, 256, ldsBytes, stream>>>(x, cb, ws, out);
}
